// FermiNetOrbitalLayer_58471684768197
// MI455X (gfx1250) — matrix operand order repaired and checked by reading, not compiled, not run on hardware
//
#include <hip/hip_runtime.h>

// FermiNet orbital layer for MI455X (gfx1250, wave32).
// out[s][b][e][o] = (x[b,s*16+e,:]·w_lin[s,:,o] + b_lin[s,o]) *
//                   sum_i w_ion[s,i] * exp(-|| r_ei[b,s*16+e,i,:] · k_env[s,i,:,:,o] ||)
//
// One wave32 handles one (b,s) tile: 16 electrons x 16 orbitals, K=256 via
// 8x v_wmma_f32_16x16x32_f16 (f16 inputs, f32 accumulate).
// Roofline: ~155 MB traffic -> ~6.6 us @ 23.3 TB/s; f16 WMMA keeps the GEMM
// far under the memory floor, envelope is VALU (v_sqrt_f32 + v_exp_f32).

typedef __attribute__((ext_vector_type(16))) _Float16 v16h;
typedef __attribute__((ext_vector_type(8)))  float    v8f;
typedef __attribute__((ext_vector_type(4)))  float    v4f;

#define NB    4096
#define NE    32
#define DDIM  256
#define NION  8
#define NORB  16
#define NSPIN 2
#define NES   16          // electrons per spin
#define KSTEPS (DDIM / 32)

__global__ __launch_bounds__(256)
void ferminet_orbital_kernel(const float* __restrict__ x,
                             const float* __restrict__ r_ei,
                             const float* __restrict__ w_lin,
                             const float* __restrict__ b_lin,
                             const float* __restrict__ k_env,
                             const float* __restrict__ w_ion,
                             float* __restrict__ out)
{
    const int lane = threadIdx.x & 31;
    const int wave = threadIdx.x >> 5;
    const int tile = blockIdx.x * 8 + wave;      // 8192 tiles = NB * NSPIN
    const int b    = tile >> 1;
    const int s    = tile & 1;

    const int g = lane >> 4;                     // half-wave select
    const int r = lane & 15;                     // A row / B,C column index

    // ---------------- envelope (VALU): env[v] for rows e = 8g+v, col o = r ---
    float env[8];
#pragma unroll
    for (int v = 0; v < 8; ++v) env[v] = 0.0f;

#pragma unroll
    for (int i = 0; i < NION; ++i) {
        // k_env[s][i][k][d][o] ; o = r for this lane (coalesced 64B rows)
        const float* kp = k_env + ((size_t)(s * NION + i) * 9) * NORB + r;
        float A00 = kp[0 * NORB], A01 = kp[1 * NORB], A02 = kp[2 * NORB];
        float A10 = kp[3 * NORB], A11 = kp[4 * NORB], A12 = kp[5 * NORB];
        float A20 = kp[6 * NORB], A21 = kp[7 * NORB], A22 = kp[8 * NORB];
        const float wi = w_ion[s * NION + i];

#pragma unroll
        for (int v = 0; v < 8; ++v) {
            const int e = 8 * g + v;
            const float* rp = r_ei +
                (((size_t)(b * NE + s * NES + e)) * NION + i) * 3;
            float r0 = rp[0], r1 = rp[1], r2 = rp[2];
            float sd0 = r0 * A00 + r1 * A10 + r2 * A20;
            float sd1 = r0 * A01 + r1 * A11 + r2 * A21;
            float sd2 = r0 * A02 + r1 * A12 + r2 * A22;
            float dist = __builtin_amdgcn_sqrtf(sd0 * sd0 + sd1 * sd1 + sd2 * sd2);
            env[v] = __builtin_fmaf(wi, __expf(-dist), env[v]);
        }
    }

    // ---------------- B fragments: w_lin[s] (256x16) -> f16, 8 K-steps -------
    v16h bfrag[KSTEPS];
    {
        const float* wp = w_lin + (size_t)s * DDIM * NORB + r;   // column r
#pragma unroll
        for (int step = 0; step < KSTEPS; ++step) {
            const int kb = step * 32 + 8 * g;
#pragma unroll
            for (int j = 0; j < 16; ++j)
                bfrag[step][j] = (_Float16)wp[(size_t)(kb + (j & 7) + ((j >> 3) << 4)) * NORB];
        }
    }

    // ---------------- GEMM: acc = x_tile * w_lin[s] (f16 WMMA, f32 acc) ------
    const float* xrow = x + (size_t)(b * NE + s * NES + r) * DDIM;
    v8f acc = {};

#pragma unroll
    for (int step = 0; step < KSTEPS; ++step) {
        const int k0 = step * 32;
        const float* p0 = xrow + k0 + 8 * g;        // halves 0..7  : K=k0+8g+j
        const float* p1 = xrow + k0 + 16 + 8 * g;   // halves 8..15 : K=k0+16+8g+j
        v4f a0 = *(const v4f*)(p0);
        v4f a1 = *(const v4f*)(p0 + 4);
        v4f a2 = *(const v4f*)(p1);
        v4f a3 = *(const v4f*)(p1 + 4);
        v16h afrag;
#pragma unroll
        for (int j = 0; j < 4; ++j) {
            afrag[j]      = (_Float16)a0[j];
            afrag[4 + j]  = (_Float16)a1[j];
            afrag[8 + j]  = (_Float16)a2[j];
            afrag[12 + j] = (_Float16)a3[j];
        }
        acc = __builtin_amdgcn_wmma_f32_16x16x32_f16(
                  /*neg_a=*/false, afrag,
                  /*neg_b=*/false, bfrag[step],
                  /*c_mod=*/(short)0, acc,
                  /*reuse_a=*/false, /*reuse_b=*/false);
    }

    // ---------------- combine + store: out[s][b][e][o] -----------------------
    const float bl = b_lin[s * NORB + r];
    float* op = out + (((size_t)s * NB + b) * NES) * NORB;
#pragma unroll
    for (int v = 0; v < 8; ++v) {
        const int e = 8 * g + v;
        op[e * NORB + r] = (acc[v] + bl) * env[v];
    }
}

extern "C" void kernel_launch(void* const* d_in, const int* in_sizes, int n_in,
                              void* d_out, int out_size, void* d_ws, size_t ws_size,
                              hipStream_t stream) {
    (void)in_sizes; (void)n_in; (void)out_size; (void)d_ws; (void)ws_size;
    const float* x     = (const float*)d_in[0];
    const float* r_ei  = (const float*)d_in[1];
    const float* w_lin = (const float*)d_in[2];
    const float* b_lin = (const float*)d_in[3];
    const float* k_env = (const float*)d_in[4];
    const float* w_ion = (const float*)d_in[5];
    float* out = (float*)d_out;

    const int tiles  = NB * NSPIN;        // 8192 (b,s) tiles, one wave each
    const int blocks = tiles / 8;         // 8 waves (256 threads) per block
    ferminet_orbital_kernel<<<blocks, 256, 0, stream>>>(
        x, r_ei, w_lin, b_lin, k_env, w_ion, out);
}
